// Trittention_52587579572809
// MI455X (gfx1250) — matrix operand order repaired and checked by reading, not compiled, not run on hardware
//
#include <hip/hip_runtime.h>
#include <stdint.h>

// ---------------------------------------------------------------------------
// Trittention for MI455X (gfx1250, wave32, WMMA bf16 + async global->LDS)
//   B=2, S=2048, D=1024, H=16, DH=64
// ---------------------------------------------------------------------------

#define BB   2
#define SS   2048
#define DD   1024
#define HH   16
#define DH   64

typedef __bf16  v16bf __attribute__((ext_vector_type(16)));
typedef __bf16  v8bf  __attribute__((ext_vector_type(8)));
typedef __bf16  v4bf  __attribute__((ext_vector_type(4)));
typedef float   v8f   __attribute__((ext_vector_type(8)));
typedef float   v4f   __attribute__((ext_vector_type(4)));

__device__ __forceinline__ v8f wmma_bf16(v16bf a, v16bf b, v8f c) {
  return __builtin_amdgcn_wmma_f32_16x16x32_bf16(
      false, a, false, b, (short)0, c, false, false);
}

// ---- CDNA5 async global->LDS copy (16B per lane), tracked by ASYNCcnt ----
__device__ __forceinline__ uint32_t lds_off(const void* p) {
  // generic pointer to LDS: low 32 bits are the wave-relative LDS byte offset
  return (uint32_t)(uintptr_t)p;
}
__device__ __forceinline__ void async_b128(void* lds_dst, const void* gsrc) {
  asm volatile("global_load_async_to_lds_b128 %0, %1, off"
               :: "v"(lds_off(lds_dst)),
                  "v"((unsigned long long)(uintptr_t)gsrc)
               : "memory");
}
__device__ __forceinline__ void wait_async0() {
  asm volatile("s_wait_asynccnt 0x0" ::: "memory");
}
__device__ __forceinline__ void wait_async4() {
  asm volatile("s_wait_asynccnt 0x4" ::: "memory");
}

// A fragment (16x32 bf16) from LDS [rows][stride].
// Lanes 0-15: row M=lane, K {0..7,16..23}; lanes 16-31: row M=lane-16, K {8..15,24..31}.
__device__ __forceinline__ v16bf frag_a(const __bf16* base, int stride,
                                        int r0, int kOff, int lane) {
  int lr = lane & 15, hi = (lane >> 4) & 1;
  const __bf16* p = base + (r0 + lr) * stride + kOff + hi * 8;
  v8bf lo  = *(const v8bf*)(p);
  v8bf hi8 = *(const v8bf*)(p + 16);
  return __builtin_shufflevector(lo, hi8, 0,1,2,3,4,5,6,7,8,9,10,11,12,13,14,15);
}
// B fragment (32x16 bf16) from LDS stored K-transposed Bs[n][k].
// Lanes 0-15: col N=lane, K {0..7,16..23}; lanes 16-31: K {8..15,24..31}.
__device__ __forceinline__ v16bf frag_b(const __bf16* base, int stride,
                                        int c0, int kOff, int lane) {
  int lr = lane & 15, hi = (lane >> 4) & 1;
  const __bf16* p = base + (c0 + lr) * stride + kOff + hi * 8;
  v8bf lo  = *(const v8bf*)(p);
  v8bf hi8 = *(const v8bf*)(p + 16);
  return __builtin_shufflevector(lo, hi8, 0,1,2,3,4,5,6,7,8,9,10,11,12,13,14,15);
}

// ---------------------------------------------------------------------------
// Prep 1: f32 -> bf16 flat convert (for hidden_states)
// ---------------------------------------------------------------------------
__global__ __launch_bounds__(256) void cvt_kernel(const float* __restrict__ X,
                                                  __bf16* __restrict__ Xb) {
  int i = (blockIdx.x * 256 + threadIdx.x) * 4;
  v4f v = *(const v4f*)(X + i);
  v4bf o; o[0]=(__bf16)v[0]; o[1]=(__bf16)v[1]; o[2]=(__bf16)v[2]; o[3]=(__bf16)v[3];
  *(v4bf*)(Xb + i) = o;
}

// ---------------------------------------------------------------------------
// Prep 2: W [K][N] f32 -> Wt [N][K] bf16 (32x32 LDS tile transpose)
// ---------------------------------------------------------------------------
__global__ __launch_bounds__(256) void wtr_kernel(const float* __restrict__ W,
                                                  __bf16* __restrict__ Wt) {
  __shared__ __bf16 T[32][36];
  const int tid = threadIdx.x;
  const int k0 = blockIdx.y * 32, n0 = blockIdx.x * 32;
  {
    int kk = tid >> 3, nn = (tid & 7) * 4;
    v4f v = *(const v4f*)(W + (size_t)(k0 + kk) * DD + n0 + nn);
#pragma unroll
    for (int j = 0; j < 4; ++j) T[nn + j][kk] = (__bf16)v[j];
  }
  __syncthreads();
  {
    int nn = tid >> 3, kk = (tid & 7) * 4;
    v4bf o = *(const v4bf*)(&T[nn][kk]);
    *(v4bf*)(Wt + (size_t)(n0 + nn) * DD + k0 + kk) = o;
  }
}

// ---------------------------------------------------------------------------
// GEMM  Y = X @ W + bias  (X bf16 [M][K] row-major, Wt bf16 [N][K])
//   OUT_MODE 0: bf16 head-split        out[b][h][s][dh]
//   OUT_MODE 2: bf16 head-split TRANS  out[b][h][dh][s]   (for k2, v1, v2)
//   OUT_MODE 1: f32 flat               out[m*N + n]
// 128x128x32 block, 8 waves (4Mx2N), double-buffered async LDS staging.
// ---------------------------------------------------------------------------
template <int OUT_MODE>
__global__ __launch_bounds__(256) void gemm_kernel(const __bf16* __restrict__ X,
                                                   const __bf16* __restrict__ Wt,
                                                   const float* __restrict__ bias,
                                                   void* __restrict__ out,
                                                   int M, int N, int K) {
  __shared__ alignas(64) __bf16 As[2][128 * 32];  // [m][k]
  __shared__ alignas(64) __bf16 Bs[2][128 * 32];  // [n][k]

  const int tid  = threadIdx.x;
  const int lane = tid & 31;
  const int wave = tid >> 5;
  const int wm = wave >> 1, wn = wave & 1;
  const int m0 = blockIdx.y * 128;
  const int n0 = blockIdx.x * 128;

  // async staging: 128 rows x 64B per tile; 256 lanes x 16B => 2 issues/matrix
  const int sr  = tid >> 2;            // 0..63
  const int sg  = (tid & 3) * 8;       // half offset within 32-half row

  v8f acc[2][4];
#pragma unroll
  for (int i = 0; i < 2; ++i)
#pragma unroll
    for (int j = 0; j < 4; ++j) acc[i][j] = (v8f)0.0f;

  // prologue: stage tile 0
  {
    const int k0 = 0;
    async_b128(&As[0][sr * 32 + sg],        X  + (size_t)(m0 + sr) * K + k0 + sg);
    async_b128(&As[0][(sr + 64) * 32 + sg], X  + (size_t)(m0 + sr + 64) * K + k0 + sg);
    async_b128(&Bs[0][sr * 32 + sg],        Wt + (size_t)(n0 + sr) * K + k0 + sg);
    async_b128(&Bs[0][(sr + 64) * 32 + sg], Wt + (size_t)(n0 + sr + 64) * K + k0 + sg);
  }

  const int nk = K / 32;
  for (int i = 0; i < nk; ++i) {
    const int buf = i & 1;
    if (i + 1 < nk) {
      const int k0 = (i + 1) * 32;
      const int nb = buf ^ 1;
      async_b128(&As[nb][sr * 32 + sg],        X  + (size_t)(m0 + sr) * K + k0 + sg);
      async_b128(&As[nb][(sr + 64) * 32 + sg], X  + (size_t)(m0 + sr + 64) * K + k0 + sg);
      async_b128(&Bs[nb][sr * 32 + sg],        Wt + (size_t)(n0 + sr) * K + k0 + sg);
      async_b128(&Bs[nb][(sr + 64) * 32 + sg], Wt + (size_t)(n0 + sr + 64) * K + k0 + sg);
      wait_async4();                    // tile i complete (tile i+1 outstanding)
    } else {
      wait_async0();
    }
    __syncthreads();                    // tile i visible to all waves

    v16bf af[2], bf[4];
#pragma unroll
    for (int mi = 0; mi < 2; ++mi)
      af[mi] = frag_a(&As[buf][0], 32, wm * 32 + mi * 16, 0, lane);
#pragma unroll
    for (int ni = 0; ni < 4; ++ni)
      bf[ni] = frag_b(&Bs[buf][0], 32, wn * 64 + ni * 16, 0, lane);
#pragma unroll
    for (int mi = 0; mi < 2; ++mi)
#pragma unroll
      for (int ni = 0; ni < 4; ++ni)
        acc[mi][ni] = wmma_bf16(af[mi], bf[ni], acc[mi][ni]);

    __syncthreads();                    // all reads of buf done before reuse
  }

  // epilogue
  const int lr = lane & 15, hi = (lane >> 4) & 1;
#pragma unroll
  for (int ni = 0; ni < 4; ++ni) {
    int gn = n0 + wn * 64 + ni * 16 + lr;
    float bv = bias[gn];
#pragma unroll
    for (int mi = 0; mi < 2; ++mi) {
      int gmb = m0 + wm * 32 + mi * 16 + hi * 8;   // first of 8 consecutive rows
      if (OUT_MODE == 2) {
        // transposed head-split: rows are contiguous -> one 16B store
        int bI = gmb >> 11, s = gmb & (SS - 1);
        int h = gn >> 6, dh = gn & (DH - 1);
        v8bf o;
#pragma unroll
        for (int r = 0; r < 8; ++r) o[r] = (__bf16)(acc[mi][ni][r] + bv);
        *(v8bf*)((__bf16*)out +
                 (((size_t)bI * HH + h) * DH + dh) * SS + s) = o;
      } else {
#pragma unroll
        for (int r = 0; r < 8; ++r) {
          int gm = gmb + r;
          float val = acc[mi][ni][r] + bv;
          if (OUT_MODE == 0) {
            int bI = gm >> 11, s = gm & (SS - 1);
            int h = gn >> 6, dh = gn & (DH - 1);
            ((__bf16*)out)[(((size_t)bI * HH + h) * SS + s) * DH + dh] = (__bf16)val;
          } else {
            ((float*)out)[(size_t)gm * N + gn] = val;
          }
        }
      }
    }
  }
}

// ---------------------------------------------------------------------------
// Kernel 2: per-head Mt = (k2^T v2)^T, inputs already [b][h][dh][s] bf16.
//   Mt[b][h][j][i] = sum_s k2[s][i] v2[s][j]  -> stored transposed for attn B-frags
// ---------------------------------------------------------------------------
__global__ __launch_bounds__(256) void kv_kernel(const __bf16* __restrict__ k2t,
                                                 const __bf16* __restrict__ v2t,
                                                 __bf16* __restrict__ Mt) {
  __shared__ alignas(64) __bf16 k2T[64 * 32];  // [dh_i][s]
  __shared__ alignas(64) __bf16 v2T[64 * 32];  // [dh_j][s]
  const int bh = blockIdx.x;
  const __bf16* K2 = k2t + (size_t)bh * DH * SS;
  const __bf16* V2 = v2t + (size_t)bh * DH * SS;
  const int tid = threadIdx.x, lane = tid & 31, wave = tid >> 5;
  const int sr = tid >> 2, sg = (tid & 3) * 8;   // 64 rows x 4 segs of 8 halves

  v8f acc[2] = {(v8f)0.0f, (v8f)0.0f};

  for (int s0 = 0; s0 < SS; s0 += 32) {
    __syncthreads();
    async_b128(&k2T[sr * 32 + sg], K2 + (size_t)sr * SS + s0 + sg);
    async_b128(&v2T[sr * 32 + sg], V2 + (size_t)sr * SS + s0 + sg);
    wait_async0();
    __syncthreads();
#pragma unroll
    for (int j = 0; j < 2; ++j) {
      int t = wave * 2 + j, mi = t >> 2, ni = t & 3;
      v16bf a = frag_a(k2T, 32, mi * 16, 0, lane);
      v16bf b = frag_b(v2T, 32, ni * 16, 0, lane);
      acc[j] = wmma_bf16(a, b, acc[j]);
    }
  }

  const int lr = lane & 15, hi = (lane >> 4) & 1;
  __bf16* Mo = Mt + (size_t)bh * DH * DH;
#pragma unroll
  for (int j = 0; j < 2; ++j) {
    int t = wave * 2 + j, mi = t >> 2, ni = t & 3;
    v8bf o;
#pragma unroll
    for (int r = 0; r < 8; ++r) o[r] = (__bf16)acc[j][r];
    // Mt[j_col][i_row]: 8 consecutive i per lane -> one 16B store
    *(v8bf*)(Mo + (size_t)(ni * 16 + lr) * DH + mi * 16 + hi * 8) = o;
  }
}

// ---------------------------------------------------------------------------
// Kernel 3: fused causal flash-attention + (O @ M) per (b,h)
//   q,k1: [b][h][s][dh];  v1t: [b][h][dh][s];  Mt: [b][h][j][i]
//   ctx out: [b][s][h*64+dh] bf16
// ---------------------------------------------------------------------------
__global__ __launch_bounds__(256) void attn_kernel(const __bf16* __restrict__ q,
                                                   const __bf16* __restrict__ k1,
                                                   const __bf16* __restrict__ v1t,
                                                   const __bf16* __restrict__ Mtg,
                                                   __bf16* __restrict__ ctx) {
  __shared__ alignas(64) __bf16 Qs[128 * 64];     // [q][dh]
  __shared__ alignas(64) __bf16 K1s[64 * 64];     // [key][dh]
  __shared__ alignas(64) __bf16 V1T[64 * 64];     // [dh][key]
  __shared__ alignas(64) __bf16 MT[64 * 64];      // [j][i]
  __shared__ alignas(64) __bf16 Ps[8 * 16 * 64];  // per-wave P / O staging

  const int bh = blockIdx.y;
  const int bI = bh >> 4, h = bh & (HH - 1);
  const int q0 = blockIdx.x * 128;
  const __bf16* Q  = q   + (size_t)bh * SS * DH;
  const __bf16* K1 = k1  + (size_t)bh * SS * DH;
  const __bf16* V1 = v1t + (size_t)bh * DH * SS;
  const __bf16* Mm = Mtg + (size_t)bh * DH * DH;

  const int tid = threadIdx.x, lane = tid & 31, wave = tid >> 5;
  const int lr = lane & 15, hi = (lane >> 4) & 1;
  const int r8 = tid >> 3, s8 = (tid & 7) * 8;    // 32 rows x 8 segs (128B rows)

  // stage Q tile (4 issues) + MT (2 issues) asynchronously
#pragma unroll
  for (int i = 0; i < 4; ++i)
    async_b128(&Qs[(r8 + 32 * i) * 64 + s8],
               Q + (size_t)(q0 + r8 + 32 * i) * DH + s8);
#pragma unroll
  for (int i = 0; i < 2; ++i)
    async_b128(&MT[(r8 + 32 * i) * 64 + s8],
               Mm + (size_t)(r8 + 32 * i) * DH + s8);

  float mrow[8], lrow[8];
  v8f oacc[4];
#pragma unroll
  for (int r = 0; r < 8; ++r) { mrow[r] = -3.0e38f; lrow[r] = 0.0f; }
#pragma unroll
  for (int ni = 0; ni < 4; ++ni) oacc[ni] = (v8f)0.0f;

  __bf16* P = Ps + wave * 16 * 64;
  const float sm_scale = 0.125f;                  // 1/sqrt(DH)
  const int kbmax = (q0 + 127) >> 6;

  for (int kb = 0; kb <= kbmax; ++kb) {
    const int k0 = kb * 64;
    __syncthreads();                              // prior reads of K1s/V1T done
#pragma unroll
    for (int i = 0; i < 2; ++i) {
      async_b128(&K1s[(r8 + 32 * i) * 64 + s8],
                 K1 + (size_t)(k0 + r8 + 32 * i) * DH + s8);
      async_b128(&V1T[(r8 + 32 * i) * 64 + s8],
                 V1 + (size_t)(r8 + 32 * i) * SS + k0 + s8);
    }
    wait_async0();
    __syncthreads();

    // ---- S = Q K1^T : 4 key tiles x 2 K-steps over DH ----
    v8f sc[4];
#pragma unroll
    for (int ni = 0; ni < 4; ++ni) sc[ni] = (v8f)0.0f;
    v16bf aq0 = frag_a(Qs, 64, wave * 16, 0, lane);
    v16bf aq1 = frag_a(Qs, 64, wave * 16, 32, lane);
#pragma unroll
    for (int ni = 0; ni < 4; ++ni) {
      v16bf b0 = frag_b(K1s, 64, ni * 16, 0, lane);
      v16bf b1 = frag_b(K1s, 64, ni * 16, 32, lane);
      sc[ni] = wmma_bf16(aq0, b0, sc[ni]);
      sc[ni] = wmma_bf16(aq1, b1, sc[ni]);
    }

    // ---- online softmax (rows live in each 16-lane half) ----
#pragma unroll
    for (int r = 0; r < 8; ++r) {
      int qrow = q0 + wave * 16 + hi * 8 + r;
      float mx = -3.0e38f;
#pragma unroll
      for (int ni = 0; ni < 4; ++ni) {
        float v = sc[ni][r] * sm_scale;
        int key = k0 + ni * 16 + lr;
        if (key > qrow) v = -3.0e38f;             // causal mask
        sc[ni][r] = v;
        mx = fmaxf(mx, v);
      }
#pragma unroll
      for (int off = 1; off < 16; off <<= 1)
        mx = fmaxf(mx, __shfl_xor(mx, off, 32));
      float mnew = fmaxf(mrow[r], mx);
      float corr = __expf(mrow[r] - mnew);
      mrow[r] = mnew;
      float rsum = 0.0f;
#pragma unroll
      for (int ni = 0; ni < 4; ++ni) {
        float p = __expf(sc[ni][r] - mnew);
        sc[ni][r] = p;
        rsum += p;
      }
#pragma unroll
      for (int off = 1; off < 16; off <<= 1)
        rsum += __shfl_xor(rsum, off, 32);
      lrow[r] = lrow[r] * corr + rsum;
#pragma unroll
      for (int ni = 0; ni < 4; ++ni) oacc[ni][r] *= corr;
    }

    // ---- P -> own-wave LDS slice, then O += P @ V1 ----
#pragma unroll
    for (int ni = 0; ni < 4; ++ni)
#pragma unroll
      for (int r = 0; r < 8; ++r)
        P[(hi * 8 + r) * 64 + ni * 16 + lr] = (__bf16)sc[ni][r];

    v16bf ap0 = frag_a(P, 64, 0, 0, lane);
    v16bf ap1 = frag_a(P, 64, 0, 32, lane);
#pragma unroll
    for (int ni = 0; ni < 4; ++ni) {
      v16bf b0 = frag_b(V1T, 64, ni * 16, 0, lane);
      v16bf b1 = frag_b(V1T, 64, ni * 16, 32, lane);
      oacc[ni] = wmma_bf16(ap0, b0, oacc[ni]);
      oacc[ni] = wmma_bf16(ap1, b1, oacc[ni]);
    }
  }

  // ---- normalize, then fused ctx = O @ M ----
#pragma unroll
  for (int r = 0; r < 8; ++r) {
    float inv = 1.0f / lrow[r];
#pragma unroll
    for (int ni = 0; ni < 4; ++ni) oacc[ni][r] *= inv;
  }
#pragma unroll
  for (int ni = 0; ni < 4; ++ni)
#pragma unroll
    for (int r = 0; r < 8; ++r)
      P[(hi * 8 + r) * 64 + ni * 16 + lr] = (__bf16)oacc[ni][r];

  v8f fin[4];
#pragma unroll
  for (int ni = 0; ni < 4; ++ni) fin[ni] = (v8f)0.0f;
  v16bf ao0 = frag_a(P, 64, 0, 0, lane);
  v16bf ao1 = frag_a(P, 64, 0, 32, lane);
#pragma unroll
  for (int ni = 0; ni < 4; ++ni) {
    v16bf b0 = frag_b(MT, 64, ni * 16, 0, lane);
    v16bf b1 = frag_b(MT, 64, ni * 16, 32, lane);
    fin[ni] = wmma_bf16(ao0, b0, fin[ni]);
    fin[ni] = wmma_bf16(ao1, b1, fin[ni]);
  }

  __bf16* co = ctx + ((size_t)bI * SS) * DD + (size_t)h * DH;
#pragma unroll
  for (int ni = 0; ni < 4; ++ni)
#pragma unroll
    for (int r = 0; r < 8; ++r) {
      int s = q0 + wave * 16 + hi * 8 + r;
      co[(size_t)s * DD + ni * 16 + lr] = (__bf16)fin[ni][r];
    }
}

// ---------------------------------------------------------------------------
// Host-side launch
// ---------------------------------------------------------------------------
extern "C" void kernel_launch(void* const* d_in, const int* in_sizes, int n_in,
                              void* d_out, int out_size, void* d_ws, size_t ws_size,
                              hipStream_t stream) {
  const float* X   = (const float*)d_in[0];
  const float* Wq  = (const float*)d_in[1];  const float* bq  = (const float*)d_in[2];
  const float* Wk1 = (const float*)d_in[3];  const float* bk1 = (const float*)d_in[4];
  const float* Wk2 = (const float*)d_in[5];  const float* bk2 = (const float*)d_in[6];
  const float* Wv1 = (const float*)d_in[7];  const float* bv1 = (const float*)d_in[8];
  const float* Wv2 = (const float*)d_in[9];  const float* bv2 = (const float*)d_in[10];
  const float* Wd  = (const float*)d_in[11]; const float* bd  = (const float*)d_in[12];

  const size_t PROJ = (size_t)BB * SS * DD * sizeof(__bf16);   // 8 MiB
  const size_t WSZ  = (size_t)DD * DD * sizeof(__bf16);        // 2 MiB
  char* ws = (char*)d_ws;
  __bf16* Xb   = (__bf16*)(ws);
  __bf16* qb   = (__bf16*)(ws + 1 * PROJ);
  __bf16* k1b  = (__bf16*)(ws + 2 * PROJ);
  __bf16* k2t  = (__bf16*)(ws + 3 * PROJ);
  __bf16* v1t  = (__bf16*)(ws + 4 * PROJ);
  __bf16* v2t  = (__bf16*)(ws + 5 * PROJ);
  __bf16* ctx  = (__bf16*)(ws + 6 * PROJ);
  __bf16* Mt   = (__bf16*)(ws + 7 * PROJ);                     // 256 KiB
  __bf16* Wts  = (__bf16*)(ws + 7 * PROJ + (1u << 20));        // 6 x 2 MiB
  __bf16* WqT  = Wts + 0 * DD * DD;
  __bf16* Wk1T = Wts + 1 * (size_t)DD * DD;
  __bf16* Wk2T = Wts + 2 * (size_t)DD * DD;
  __bf16* Wv1T = Wts + 3 * (size_t)DD * DD;
  __bf16* Wv2T = Wts + 4 * (size_t)DD * DD;
  __bf16* WdT  = Wts + 5 * (size_t)DD * DD;
  (void)WSZ;

  const int M = BB * SS, N = DD, K = DD;
  dim3 thr(256);

  // 0) precision/layout prep (bandwidth-trivial vs 23.3 TB/s)
  cvt_kernel<<<(M * K) / (256 * 4), thr, 0, stream>>>(X, Xb);
  dim3 tg(DD / 32, DD / 32);
  wtr_kernel<<<tg, thr, 0, stream>>>(Wq,  WqT);
  wtr_kernel<<<tg, thr, 0, stream>>>(Wk1, Wk1T);
  wtr_kernel<<<tg, thr, 0, stream>>>(Wk2, Wk2T);
  wtr_kernel<<<tg, thr, 0, stream>>>(Wv1, Wv1T);
  wtr_kernel<<<tg, thr, 0, stream>>>(Wv2, Wv2T);
  wtr_kernel<<<tg, thr, 0, stream>>>(Wd,  WdT);

  // 1) projections (WMMA, async double-buffered LDS)
  dim3 gblk(N / 128, M / 128);
  gemm_kernel<0><<<gblk, thr, 0, stream>>>(Xb, WqT,  bq,  qb,  M, N, K);
  gemm_kernel<0><<<gblk, thr, 0, stream>>>(Xb, Wk1T, bk1, k1b, M, N, K);
  gemm_kernel<2><<<gblk, thr, 0, stream>>>(Xb, Wk2T, bk2, k2t, M, N, K);
  gemm_kernel<2><<<gblk, thr, 0, stream>>>(Xb, Wv1T, bv1, v1t, M, N, K);
  gemm_kernel<2><<<gblk, thr, 0, stream>>>(Xb, Wv2T, bv2, v2t, M, N, K);

  // 2) per-head Mt = (k2^T v2)^T  — collapses the two trailing S-scale matmuls
  kv_kernel<<<BB * HH, thr, 0, stream>>>(k2t, v2t, Mt);

  // 3) fused flash attention + O@M -> ctx (bf16, heads recombined)
  attn_kernel<<<dim3(SS / 128, BB * HH), thr, 0, stream>>>(qb, k1b, v1t, Mt, ctx);

  // 4) output projection -> f32
  gemm_kernel<1><<<gblk, thr, 0, stream>>>(ctx, WdT, bd, (float*)d_out, M, N, K);
}
